// SelfAttention_49039936586148
// MI455X (gfx1250) — compile-verified, matrix operand order checked by reading, not run on hardware
//
#include <hip/hip_runtime.h>

// ---------------------------------------------------------------------------
// Self-attention for MI455X (gfx1250, wave32, WMMA + Tensor Data Mover).
// B=2, T=2048, C=1024, H=16, D=64. All GEMMs via v_wmma_f32_16x16x32_bf16.
// V is produced pre-transposed per head ([B,H,D,T]) so the attention PV
// operand tile is a plain 2D strided TDM load (no in-loop transpose).
// ---------------------------------------------------------------------------

typedef __attribute__((ext_vector_type(16))) __bf16 bf16x16;
typedef __attribute__((ext_vector_type(8)))  __bf16 bf16x8;
typedef __attribute__((ext_vector_type(8)))  float  f32x8;
typedef __attribute__((ext_vector_type(4)))  unsigned int u32x4;
typedef __attribute__((ext_vector_type(4)))  int  i32x4;
typedef __attribute__((ext_vector_type(8)))  int  i32x8;

#define TSEQ   2048
#define CDIM   1024
#define NHEAD  16
#define HDIM   64
#define MROWS  4096   // B*T

// ---- Tensor Data Mover availability / arity -------------------------------
#if defined(__has_builtin)
#if __has_builtin(__builtin_amdgcn_tensor_load_to_lds)
#define HAVE_TDM 1
#else
#define HAVE_TDM 0
#endif
#if __has_builtin(__builtin_amdgcn_s_wait_tensorcnt)
#define HAVE_WAIT_TENSOR 1
#else
#define HAVE_WAIT_TENSOR 0
#endif
#else
#define HAVE_TDM 0
#define HAVE_WAIT_TENSOR 0
#endif

#if defined(__has_include)
#if __has_include(<hip/amd_detail/amd_gfx1250_TDM.h>)
#define TDM_SIX_ARGS 1
#else
#define TDM_SIX_ARGS 0
#endif
#else
#define TDM_SIX_ARGS 0
#endif

__device__ __forceinline__ unsigned lds_offset_of(const void* p) {
    // LDS aperture: addr[31:0] is the LDS byte offset (ISA 10.2).
    return (unsigned)(size_t)p;
}

__device__ __forceinline__ void wait_tensor0() {
#if HAVE_WAIT_TENSOR
    __builtin_amdgcn_s_wait_tensorcnt(0);
#else
    asm volatile("s_wait_tensorcnt 0" ::: "memory");
#endif
    asm volatile("" ::: "memory");
}

#if HAVE_TDM
// Issue one TDM 2D tile load: rows x cols (bf16 elements) from global
// (row stride = row_stride_elems) into LDS at lds_byte_off.
// pad16: insert 4 DWORDs of LDS padding every 16 DWORDs stored
//        (i.e. 32-elem rows land at a stride of 40 elements).
__device__ __forceinline__ void tdm_load_2d(const void* gsrc,
                                            unsigned lds_byte_off,
                                            unsigned rows, unsigned cols,
                                            unsigned row_stride_elems,
                                            bool pad16) {
    unsigned long long ga = (unsigned long long)(size_t)gsrc;
    u32x4 g0 = { 1u,                               // count=1 (valid descriptor)
                 lds_byte_off,                     // lds_addr
                 (unsigned)(ga & 0xffffffffu),     // global_addr[31:0]
                 (unsigned)((ga >> 32) & 0x1ffffffu) | (2u << 30) }; // + type=2
    unsigned d0 = 0x00010000u;                     // data_size = 2 bytes
    if (pad16) d0 |= (1u << 20)                    // pad_enable
                   | (3u << 22)                    // pad_interval: 16 DWORDs
                   | (3u << 25);                   // pad_amount: 4 DWORDs
    i32x8 g1 = { (int)d0,
                 (int)((cols & 0xffffu) << 16),                    // tdim0 lo16
                 (int)((cols >> 16) | ((rows & 0xffffu) << 16)),   // tdim0 hi | tdim1 lo
                 (int)((rows >> 16) | ((cols & 0xffffu) << 16)),   // tdim1 hi | tile_dim0
                 (int)(rows & 0xffffu),                            // tile_dim1 (tile_dim2=0)
                 (int)row_stride_elems,                            // dim0_stride lo32
                 0, 0 };
    i32x4 gz = { 0, 0, 0, 0 };
#if TDM_SIX_ARGS
    i32x8 gz8 = { 0, 0, 0, 0, 0, 0, 0, 0 };
    __builtin_amdgcn_tensor_load_to_lds(g0, g1, gz, gz, gz8, 0);
#else
    __builtin_amdgcn_tensor_load_to_lds(g0, g1, gz, gz, 0);
#endif
}
#endif // HAVE_TDM

// ---------------------------------------------------------------------------
// 16x32 bf16 fragment loader (A-layout; both GEMM operands are K-contiguous).
// ---------------------------------------------------------------------------
__device__ __forceinline__ bf16x16 load_frag(const __bf16* tile, int stride) {
    const int lane = threadIdx.x & 31;
    const __bf16* row = tile + (lane & 15) * stride;
    const int kh = (lane >> 4) * 8;
    union { bf16x16 v; bf16x8 h[2]; } u;
    u.h[0] = *reinterpret_cast<const bf16x8*>(row + kh);
    u.h[1] = *reinterpret_cast<const bf16x8*>(row + 16 + kh);
    return u.v;
}

__device__ __forceinline__ f32x8 wmma_bf16(bf16x16 a, bf16x16 b, f32x8 c) {
    return __builtin_amdgcn_wmma_f32_16x16x32_bf16(
        false, a, false, b, (short)0, c, false, false);
}

// ---------------------------------------------------------------------------
// fp32 -> bf16 conversion
// ---------------------------------------------------------------------------
__global__ void f32_to_bf16_kernel(const float* __restrict__ src,
                                   __bf16* __restrict__ dst, int n) {
    for (int i = blockIdx.x * blockDim.x + threadIdx.x; i < n;
         i += gridDim.x * blockDim.x) {
        dst[i] = (__bf16)src[i];
    }
}

// ---------------------------------------------------------------------------
// NT GEMM: out[m, n] = sum_k A[m, k] * W[n, k] + bias[n]
// Workgroup: 256 thr = 8 waves; tile 128(M) x 128(N); wave = 32x64 (2x4 frags).
// LDS double-buffered, staged by TDM (fallback: uint4 copies).
// mode 0: fp32 out[m*1024+n]
// mode 1: bf16 head-major   [B,H,T,D]  (Q, K)
// mode 2: bf16 head-transposed [B,H,D,T]  (V)
// ---------------------------------------------------------------------------
#define GSTRIDE 40   // padded LDS row stride (elements) for 32-elem rows

__global__ __launch_bounds__(256) void gemm_nt_bf16_kernel(
    const __bf16* __restrict__ A, const __bf16* __restrict__ W,
    const float* __restrict__ bias, float* __restrict__ outF,
    __bf16* __restrict__ outH, int mode) {

    __shared__ __bf16 sA[2][128 * GSTRIDE];
    __shared__ __bf16 sB[2][128 * GSTRIDE];

    const int tid  = threadIdx.x;
    const int lane = tid & 31;
    const int wave = tid >> 5;
    const int wm   = wave & 3;        // M strip: wm*32
    const int wn   = wave >> 2;       // N strip: wn*64
    const int m0   = blockIdx.x * 128;
    const int n0   = blockIdx.y * 128;

    f32x8 acc[2][4];
#pragma unroll
    for (int i = 0; i < 2; ++i)
#pragma unroll
        for (int j = 0; j < 4; ++j) acc[i][j] = (f32x8)0.0f;

    auto stage = [&](int buf, int kb) {
#if HAVE_TDM && defined(__gfx1250__)
        tdm_load_2d(&A[(m0 + wave * 16) * CDIM + kb],
                    lds_offset_of(&sA[buf][wave * 16 * GSTRIDE]),
                    16, 32, CDIM, true);
        tdm_load_2d(&W[(n0 + wave * 16) * CDIM + kb],
                    lds_offset_of(&sB[buf][wave * 16 * GSTRIDE]),
                    16, 32, CDIM, true);
#else
#pragma unroll
        for (int c = 0; c < 2; ++c) {
            int id = tid + c * 256;
            int r = id >> 2, c8 = (id & 3) * 8;
            *reinterpret_cast<uint4*>(&sA[buf][r * GSTRIDE + c8]) =
                *reinterpret_cast<const uint4*>(&A[(m0 + r) * CDIM + kb + c8]);
            *reinterpret_cast<uint4*>(&sB[buf][r * GSTRIDE + c8]) =
                *reinterpret_cast<const uint4*>(&W[(n0 + r) * CDIM + kb + c8]);
        }
#endif
    };
    auto stage_wait = [&]() {
#if HAVE_TDM && defined(__gfx1250__)
        wait_tensor0();
#endif
        __syncthreads();
    };

    stage(0, 0);
    stage_wait();

    const int KT = CDIM / 32;
    for (int kt = 0; kt < KT; ++kt) {
        const int cur = kt & 1;
        if (kt + 1 < KT) {
            __builtin_prefetch(&A[(m0 + wave * 16) * CDIM + (kt + 2) * 32], 0, 1);
            stage(cur ^ 1, (kt + 1) * 32);
        }

        bf16x16 a0 = load_frag(&sA[cur][(wm * 32) * GSTRIDE], GSTRIDE);
        bf16x16 a1 = load_frag(&sA[cur][(wm * 32 + 16) * GSTRIDE], GSTRIDE);
#pragma unroll
        for (int j = 0; j < 4; ++j) {
            bf16x16 b = load_frag(&sB[cur][(wn * 64 + j * 16) * GSTRIDE], GSTRIDE);
            acc[0][j] = wmma_bf16(a0, b, acc[0][j]);
            acc[1][j] = wmma_bf16(a1, b, acc[1][j]);
        }

        if (kt + 1 < KT) stage_wait();
    }

    // --- epilogue -----------------------------------------------------------
#pragma unroll
    for (int mi = 0; mi < 2; ++mi) {
#pragma unroll
        for (int ni = 0; ni < 4; ++ni) {
#pragma unroll
            for (int i = 0; i < 8; ++i) {
                int gm  = m0 + wm * 32 + mi * 16 + i + ((lane >> 4) << 3);
                int col = n0 + wn * 64 + ni * 16 + (lane & 15);
                float v = acc[mi][ni][i] + bias[col];
                if (mode == 0) {
                    outF[gm * CDIM + col] = v;
                } else {
                    int h = col >> 6, d = col & 63;
                    int b = gm >> 11, t = gm & 2047;
                    if (mode == 1) {
                        outH[((b * NHEAD + h) * TSEQ + t) * HDIM + d] = (__bf16)v;
                    } else {   // mode 2: per-head transposed [B,H,D,T]
                        outH[((b * NHEAD + h) * HDIM + d) * TSEQ + t] = (__bf16)v;
                    }
                }
            }
        }
    }
}

// ---------------------------------------------------------------------------
// Flash attention. Grid: 512 WGs = (B*H=32) x (T/128=16 query blocks).
// 8 waves/WG, each wave owns 16 query rows. K and V^T tiles double-buffered
// and staged by the TDM (V^T is a strided 2D tile thanks to [B,H,D,T] layout).
// ---------------------------------------------------------------------------
__global__ __launch_bounds__(256) void attn_kernel(
    const __bf16* __restrict__ Q, const __bf16* __restrict__ K,
    const __bf16* __restrict__ Vt, __bf16* __restrict__ ctx) {

    __shared__ __bf16 sK[2][64 * 64];    // [key][d]
    __shared__ __bf16 sVt[2][64 * 64];   // [d][key]
    __shared__ __bf16 sP[8 * 16 * 64];   // per-wave P staging

    const int tid  = threadIdx.x;
    const int lane = tid & 31;
    const int wave = tid >> 5;
    const int bh   = blockIdx.x >> 4;
    const int qb   = blockIdx.x & 15;
    const int b    = bh >> 4;
    const int h    = bh & 15;
    const int q0   = qb * 128 + wave * 16;

    const __bf16* Qbase = Q + (bh * TSEQ + q0) * HDIM;
    bf16x16 qf[2];
    qf[0] = load_frag(Qbase, HDIM);
    qf[1] = load_frag(Qbase + 32, HDIM);

    float m_i[8], l_i[8];
    f32x8 o[4];
#pragma unroll
    for (int i = 0; i < 8; ++i) { m_i[i] = -1.0e30f; l_i[i] = 0.0f; }
#pragma unroll
    for (int j = 0; j < 4; ++j) o[j] = (f32x8)0.0f;

    const int kend = qb * 128 + 128;
    __bf16* Pw = sP + wave * 16 * 64;

    auto stage = [&](int buf, int kb) {
        const __bf16* Kb  = K  + (bh * TSEQ + kb) * HDIM;       // rows: keys
        const __bf16* Vtb = Vt + bh * HDIM * TSEQ + kb;         // rows: d
#if HAVE_TDM && defined(__gfx1250__)
        // wave-split slices: 8 rows each
        tdm_load_2d(Kb + wave * 8 * HDIM,
                    lds_offset_of(&sK[buf][wave * 8 * 64]),
                    8, 64, HDIM, false);
        tdm_load_2d(Vtb + wave * 8 * TSEQ,
                    lds_offset_of(&sVt[buf][wave * 8 * 64]),
                    8, 64, TSEQ, false);
#else
#pragma unroll
        for (int c = 0; c < 2; ++c) {
            int id = tid + c * 256;                  // 512 x 16B chunks
            reinterpret_cast<uint4*>(&sK[buf][0])[id] =
                reinterpret_cast<const uint4*>(Kb)[id];
            int r = id >> 3, c8 = (id & 7) * 8;
            *reinterpret_cast<uint4*>(&sVt[buf][r * 64 + c8]) =
                *reinterpret_cast<const uint4*>(&Vtb[r * TSEQ + c8]);
        }
#endif
    };

    stage(0, 0);
#if HAVE_TDM && defined(__gfx1250__)
    wait_tensor0();
#endif
    __syncthreads();

    for (int kb = 0; kb < kend; kb += 64) {
        const int cur = (kb >> 6) & 1;
        const bool has_next = (kb + 64 < kend);
        if (has_next) stage(cur ^ 1, kb + 64);

        // --- S = Q K^T ---
        f32x8 s[4];
#pragma unroll
        for (int j = 0; j < 4; ++j) s[j] = (f32x8)0.0f;
#pragma unroll
        for (int c = 0; c < 2; ++c) {
#pragma unroll
            for (int j = 0; j < 4; ++j) {
                bf16x16 bf = load_frag(&sK[cur][(j * 16) * 64 + c * 32], 64);
                s[j] = wmma_bf16(qf[c], bf, s[j]);
            }
        }

        // --- scale + causal mask + row max ---
        float rmax[8];
#pragma unroll
        for (int i = 0; i < 8; ++i) rmax[i] = -1.0e30f;
#pragma unroll
        for (int j = 0; j < 4; ++j) {
#pragma unroll
            for (int i = 0; i < 8; ++i) {
                int kg = kb + j * 16 + (lane & 15);
                int qg = q0 + i + ((lane >> 4) << 3);
                float sv = s[j][i] * 0.125f;
                if (kg > qg) sv = -1.0e30f;
                s[j][i] = sv;
                rmax[i] = fmaxf(rmax[i], sv);
            }
        }
#pragma unroll
        for (int off = 1; off < 16; off <<= 1)
#pragma unroll
            for (int i = 0; i < 8; ++i)
                rmax[i] = fmaxf(rmax[i], __shfl_xor(rmax[i], off, 32));

        // --- online softmax ---
        float alpha[8], rsum[8];
#pragma unroll
        for (int i = 0; i < 8; ++i) {
            float mn = fmaxf(m_i[i], rmax[i]);
            alpha[i] = __expf(m_i[i] - mn);
            m_i[i] = mn;
            rsum[i] = 0.0f;
        }
#pragma unroll
        for (int j = 0; j < 4; ++j) {
#pragma unroll
            for (int i = 0; i < 8; ++i) {
                float p = __expf(s[j][i] - m_i[i]);
                rsum[i] += p;
                Pw[(i + ((lane >> 4) << 3)) * 64 + j * 16 + (lane & 15)] =
                    (__bf16)p;
            }
        }
#pragma unroll
        for (int off = 1; off < 16; off <<= 1)
#pragma unroll
            for (int i = 0; i < 8; ++i)
                rsum[i] += __shfl_xor(rsum[i], off, 32);
#pragma unroll
        for (int i = 0; i < 8; ++i) l_i[i] = l_i[i] * alpha[i] + rsum[i];
#pragma unroll
        for (int j = 0; j < 4; ++j)
#pragma unroll
            for (int i = 0; i < 8; ++i) o[j][i] *= alpha[i];

        asm volatile("s_wait_dscnt 0" ::: "memory");

        // --- O += P V ---
#pragma unroll
        for (int c = 0; c < 2; ++c) {
            bf16x16 pf = load_frag(Pw + c * 32, 64);
#pragma unroll
            for (int j = 0; j < 4; ++j) {
                bf16x16 vf = load_frag(&sVt[cur][(j * 16) * 64 + c * 32], 64);
                o[j] = wmma_bf16(pf, vf, o[j]);
            }
        }

#if HAVE_TDM && defined(__gfx1250__)
        if (has_next) wait_tensor0();
#endif
        __syncthreads();
    }

    // --- normalize + write context ---
#pragma unroll
    for (int j = 0; j < 4; ++j) {
#pragma unroll
        for (int i = 0; i < 8; ++i) {
            int qg  = q0 + i + ((lane >> 4) << 3);
            int col = h * HDIM + j * 16 + (lane & 15);
            float v = o[j][i] * (1.0f / l_i[i]);
            ctx[(b * TSEQ + qg) * CDIM + col] = (__bf16)v;
        }
    }
}

// ---------------------------------------------------------------------------
// Launcher
// ---------------------------------------------------------------------------
extern "C" void kernel_launch(void* const* d_in, const int* in_sizes, int n_in,
                              void* d_out, int out_size, void* d_ws, size_t ws_size,
                              hipStream_t stream) {
    const float* x  = (const float*)d_in[0];
    const float* Wq = (const float*)d_in[1];
    const float* bq = (const float*)d_in[2];
    const float* Wk = (const float*)d_in[3];
    const float* bk = (const float*)d_in[4];
    const float* Wv = (const float*)d_in[5];
    const float* bv = (const float*)d_in[6];
    const float* Wo = (const float*)d_in[7];
    const float* bo = (const float*)d_in[8];

    char* ws = (char*)d_ws;
    size_t off = 0;
    __bf16* xb  = (__bf16*)(ws + off); off += (size_t)MROWS * CDIM * 2;
    __bf16* wqb = (__bf16*)(ws + off); off += (size_t)CDIM * CDIM * 2;
    __bf16* wkb = (__bf16*)(ws + off); off += (size_t)CDIM * CDIM * 2;
    __bf16* wvb = (__bf16*)(ws + off); off += (size_t)CDIM * CDIM * 2;
    __bf16* wob = (__bf16*)(ws + off); off += (size_t)CDIM * CDIM * 2;
    __bf16* Qp  = (__bf16*)(ws + off); off += (size_t)MROWS * CDIM * 2;  // [B,H,T,D]
    __bf16* Kp  = (__bf16*)(ws + off); off += (size_t)MROWS * CDIM * 2;  // [B,H,T,D]
    __bf16* Vtp = (__bf16*)(ws + off); off += (size_t)MROWS * CDIM * 2;  // [B,H,D,T]
    __bf16* ctx = (__bf16*)(ws + off); off += (size_t)MROWS * CDIM * 2;

    f32_to_bf16_kernel<<<2048, 256, 0, stream>>>(x,  xb,  MROWS * CDIM);
    f32_to_bf16_kernel<<<1024, 256, 0, stream>>>(Wq, wqb, CDIM * CDIM);
    f32_to_bf16_kernel<<<1024, 256, 0, stream>>>(Wk, wkb, CDIM * CDIM);
    f32_to_bf16_kernel<<<1024, 256, 0, stream>>>(Wv, wvb, CDIM * CDIM);
    f32_to_bf16_kernel<<<1024, 256, 0, stream>>>(Wo, wob, CDIM * CDIM);

    dim3 gg(MROWS / 128, CDIM / 128, 1);
    gemm_nt_bf16_kernel<<<gg, 256, 0, stream>>>(xb, wqb, bq, nullptr, Qp,  1);
    gemm_nt_bf16_kernel<<<gg, 256, 0, stream>>>(xb, wkb, bk, nullptr, Kp,  1);
    gemm_nt_bf16_kernel<<<gg, 256, 0, stream>>>(xb, wvb, bv, nullptr, Vtp, 2);

    attn_kernel<<<512, 256, 0, stream>>>(Qp, Kp, Vtp, ctx);

    gemm_nt_bf16_kernel<<<gg, 256, 0, stream>>>(ctx, wob, bo, (float*)d_out,
                                                nullptr, 0);
}
